// GATLayer_68058051772587
// MI455X (gfx1250) — compile-verified, matrix operand order checked by reading, not run on hardware
//
#include <hip/hip_runtime.h>
#include <hip/hip_bf16.h>

typedef __bf16 bf16;
typedef __bf16 v16bf __attribute__((ext_vector_type(16)));
typedef __bf16 v8bf  __attribute__((ext_vector_type(8)));
typedef float  v8f   __attribute__((ext_vector_type(8)));

#define N_NODES 4096
#define F_IN    256
#define FH      256   // F_OUT * H
#define F_OUT   64
#define NHEAD   4
#define ALPHA   (-0.2f)
#define CAP     1024  // max neighbors kept per row (p=0.015 -> mean ~61, P(deg>1024)~0)

// ---------------- precision conversion ----------------
__global__ __launch_bounds__(256) void k_f32_to_bf16(const float* __restrict__ x,
                                                     bf16* __restrict__ y, int n) {
  int i = blockIdx.x * 256 + threadIdx.x;
  if (i < n) y[i] = (bf16)x[i];
}

// y[c*rows + r] = x[r*cols + c]  (store B transposed so WMMA B-frag loads are contiguous)
__global__ __launch_bounds__(256) void k_f32_to_bf16_T(const float* __restrict__ x,
                                                       bf16* __restrict__ y, int rows, int cols) {
  int i = blockIdx.x * 256 + threadIdx.x;
  if (i < rows * cols) {
    int r = i / cols, c = i % cols;
    y[c * rows + r] = (bf16)x[i];
  }
}

// ---------------- WMMA GEMM: C[M,N] = A[M,K] * B, with B stored as BT[N,K] ----------------
// One 16x16 output tile per wave, K-loop in steps of 32 using v_wmma_f32_16x16x32_bf16.
__global__ __launch_bounds__(256) void k_wmma_gemm_bf16(const bf16* __restrict__ A,
                                                        const bf16* __restrict__ BT,
                                                        float* __restrict__ C,
                                                        int M, int N, int K) {
  int wid  = blockIdx.x * 8 + (threadIdx.x >> 5);   // global wave id (8 waves / block)
  int lane = threadIdx.x & 31;
  int ntile = N >> 4;
  int tm = wid / ntile, tn = wid % ntile;
  if (tm * 16 >= M) return;
  int l15  = lane & 15;
  int half = lane >> 4;
  // A frag: row = tm*16 + l15, k = half*8 + {0..7} and {16..23}  (ISA 16-bit A layout)
  const bf16* arow = A  + (size_t)(tm * 16 + l15) * K + half * 8;
  // B frag: col = tn*16 + l15, same k pattern; contiguous in BT[N,K]
  const bf16* brow = BT + (size_t)(tn * 16 + l15) * K + half * 8;
  v8f acc = {};
  for (int k0 = 0; k0 < K; k0 += 32) {
    v8bf alo = *(const v8bf*)(arow + k0);
    v8bf ahi = *(const v8bf*)(arow + k0 + 16);
    v8bf blo = *(const v8bf*)(brow + k0);
    v8bf bhi = *(const v8bf*)(brow + k0 + 16);
    v16bf av = __builtin_shufflevector(alo, ahi, 0,1,2,3,4,5,6,7,8,9,10,11,12,13,14,15);
    v16bf bv = __builtin_shufflevector(blo, bhi, 0,1,2,3,4,5,6,7,8,9,10,11,12,13,14,15);
    acc = __builtin_amdgcn_wmma_f32_16x16x32_bf16(false, av, false, bv, (short)0, acc,
                                                  false, false);
  }
  // D layout: lane holds col l15, rows half*8 + v
  float* crow = C + (size_t)(tm * 16 + half * 8) * N + tn * 16 + l15;
#pragma unroll
  for (int v = 0; v < 8; ++v) crow[(size_t)v * N] = acc[v];
}

// ---------------- per-node attention scalars: s_src/s_dst[n,h] ----------------
__global__ __launch_bounds__(256) void k_scores(const float* __restrict__ h,
                                                const float* __restrict__ a,
                                                float* __restrict__ ssrc,
                                                float* __restrict__ sdst) {
  int idx = blockIdx.x * 256 + threadIdx.x;   // n*4 + head
  if (idx >= N_NODES * NHEAD) return;
  int n = idx >> 2, hh = idx & 3;
  const float* hp = h + (size_t)n * FH + hh;  // hr[n,f,hh] = h[n, f*4+hh]
  float s1 = 0.f, s2 = 0.f;
  for (int f = 0; f < F_OUT; ++f) {
    float v = hp[f * 4];
    s1 += v * a[f * 4 + hh];
    s2 += v * a[(F_OUT + f) * 4 + hh];
  }
  ssrc[idx] = s1;
  sdst[idx] = s2;
}

// ---------------- colsum[c] = sum_n h[n,c] ----------------
__global__ __launch_bounds__(256) void k_colsum(const float* __restrict__ h,
                                                float* __restrict__ cs) {
  __shared__ float red[256];
  int c = blockIdx.x, t = threadIdx.x;
  float s = 0.f;
  for (int n = t; n < N_NODES; n += 256) s += h[(size_t)n * FH + c];
  red[t] = s;
  __syncthreads();
  for (int w = 128; w > 0; w >>= 1) {
    if (t < w) red[t] += red[t + w];
    __syncthreads();
  }
  if (t == 0) cs[c] = red[0];
}

// ---------------- fused attention row kernel ----------------
// Exploits binary adj * score: non-neighbor softmax logits are exactly 0.
__global__ __launch_bounds__(256) void k_gat_attn(const float* __restrict__ adj,
                                                  const float* __restrict__ h,
                                                  const float* __restrict__ ssrc,
                                                  const float* __restrict__ sdst,
                                                  const float* __restrict__ colsum,
                                                  const float* __restrict__ resid,
                                                  const float* __restrict__ bias,
                                                  const float* __restrict__ res_b,
                                                  float* __restrict__ out) {
  __shared__ int   s_cnt[256];
  __shared__ int   s_idx[CAP];
  __shared__ float s_w[CAP * NHEAD];
  __shared__ float s_red[256 * NHEAD];
  __shared__ float s_m[NHEAD];
  __shared__ float s_d[NHEAD];
  __shared__ int   s_deg;

  int i = blockIdx.x;
  int t = threadIdx.x;
  const float* arow = adj + (size_t)i * N_NODES;

  // read my 16 contiguous adj entries (fully coalesced 64B/lane)
  float av[16];
  {
    const float4* p = (const float4*)(arow + t * 16);
#pragma unroll
    for (int q = 0; q < 4; ++q) {
      float4 v = p[q];
      av[4*q+0] = v.x; av[4*q+1] = v.y; av[4*q+2] = v.z; av[4*q+3] = v.w;
    }
  }
  int cnt = 0;
#pragma unroll
  for (int q = 0; q < 16; ++q) cnt += (av[q] != 0.0f) ? 1 : 0;
  s_cnt[t] = cnt;
  __syncthreads();
  if (t == 0) {                      // deterministic exclusive scan
    int acc = 0;
    for (int u = 0; u < 256; ++u) { int c = s_cnt[u]; s_cnt[u] = acc; acc += c; }
    s_deg = acc;
  }
  __syncthreads();
  int degTrue = s_deg;
  int off = s_cnt[t];
#pragma unroll
  for (int q = 0; q < 16; ++q)
    if (av[q] != 0.0f) { if (off < CAP) s_idx[off] = t * 16 + q; off++; }
  __syncthreads();
  int deg = degTrue > CAP ? CAP : degTrue;

  float4 si4 = *(const float4*)(ssrc + i * 4);
  float sia[4] = {si4.x, si4.y, si4.z, si4.w};

  // pass 1: per-head max of leaky(e) over neighbors
  float lmx[4] = {-1e30f, -1e30f, -1e30f, -1e30f};
  for (int p = t; p < deg; p += 256) {
    float4 sd = *(const float4*)(sdst + s_idx[p] * 4);
    float sda[4] = {sd.x, sd.y, sd.z, sd.w};
#pragma unroll
    for (int hh = 0; hh < 4; ++hh) {
      float e = sia[hh] + sda[hh];
      e = (e >= 0.0f) ? e : ALPHA * e;
      lmx[hh] = fmaxf(lmx[hh], e);
    }
  }
#pragma unroll
  for (int hh = 0; hh < 4; ++hh) s_red[t * 4 + hh] = lmx[hh];
  __syncthreads();
  for (int w = 128; w > 0; w >>= 1) {
    if (t < w) {
#pragma unroll
      for (int hh = 0; hh < 4; ++hh)
        s_red[t * 4 + hh] = fmaxf(s_red[t * 4 + hh], s_red[(t + w) * 4 + hh]);
    }
    __syncthreads();
  }
  if (t < 4) {
    float m = s_red[t];
    if (degTrue < N_NODES) m = fmaxf(m, 0.0f);  // non-neighbors contribute logit 0
    s_m[t] = m;
  }
  __syncthreads();
  float mh[4] = {s_m[0], s_m[1], s_m[2], s_m[3]};

  // pass 2: softmax weights + denominator partials
  float lden[4] = {0.f, 0.f, 0.f, 0.f};
  for (int p = t; p < deg; p += 256) {
    float4 sd = *(const float4*)(sdst + s_idx[p] * 4);
    float sda[4] = {sd.x, sd.y, sd.z, sd.w};
#pragma unroll
    for (int hh = 0; hh < 4; ++hh) {
      float e = sia[hh] + sda[hh];
      e = (e >= 0.0f) ? e : ALPHA * e;
      float w = __expf(e - mh[hh]);
      s_w[p * 4 + hh] = w;
      lden[hh] += w;
    }
  }
  __syncthreads();
#pragma unroll
  for (int hh = 0; hh < 4; ++hh) s_red[t * 4 + hh] = lden[hh];
  __syncthreads();
  for (int w = 128; w > 0; w >>= 1) {
    if (t < w) {
#pragma unroll
      for (int hh = 0; hh < 4; ++hh)
        s_red[t * 4 + hh] += s_red[(t + w) * 4 + hh];
    }
    __syncthreads();
  }
  if (t < 4)
    s_d[t] = s_red[t] + (float)(N_NODES - degTrue) * __expf(-mh[t]);
  __syncthreads();

  // pass 3: one output column per thread (head = t>>6, f = t&63)
  int hh = t >> 6;
  int f  = t & 63;
  float em  = __expf(-mh[hh]);
  float den = s_d[hh];
  int col = f * 4 + hh;                        // hr[j,f,hh] column in h
  float acc = em * colsum[col];                // all-j background at weight e^{-m}
  for (int p = 0; p < deg; ++p)
    acc += (s_w[p * 4 + hh] - em) * h[(size_t)s_idx[p] * FH + col];
  int oc = hh * F_OUT + f;                     // out layout [N, H*F_OUT]
  size_t o = (size_t)i * FH + oc;
  out[o] = acc / den + bias[oc] + resid[o] + res_b[oc];
}

// ---------------- launch ----------------
extern "C" void kernel_launch(void* const* d_in, const int* in_sizes, int n_in,
                              void* d_out, int out_size, void* d_ws, size_t ws_size,
                              hipStream_t stream) {
  const float* inputs = (const float*)d_in[0];
  const float* adj    = (const float*)d_in[1];
  const float* W      = (const float*)d_in[2];
  const float* a      = (const float*)d_in[3];
  const float* bias   = (const float*)d_in[4];
  const float* res_w  = (const float*)d_in[5];
  const float* res_b  = (const float*)d_in[6];
  float* out = (float*)d_out;

  char* ws = (char*)d_ws;
  size_t off = 0;
  auto alloc = [&](size_t bytes) -> void* {
    void* p = ws + off;
    off += (bytes + 255) & ~(size_t)255;
    return p;
  };
  float* h     = (float*)alloc((size_t)N_NODES * FH * 4);
  float* resid = (float*)alloc((size_t)N_NODES * FH * 4);
  float* ssrc  = (float*)alloc((size_t)N_NODES * NHEAD * 4);
  float* sdst  = (float*)alloc((size_t)N_NODES * NHEAD * 4);
  float* cs    = (float*)alloc((size_t)FH * 4);
  bf16* Abf    = (bf16*)alloc((size_t)N_NODES * F_IN * 2);
  bf16* WT     = (bf16*)alloc((size_t)F_IN * FH * 2);
  bf16* RT     = (bf16*)alloc((size_t)F_IN * FH * 2);

  k_f32_to_bf16<<<(N_NODES * F_IN + 255) / 256, 256, 0, stream>>>(inputs, Abf, N_NODES * F_IN);
  k_f32_to_bf16_T<<<(F_IN * FH + 255) / 256, 256, 0, stream>>>(W, WT, F_IN, FH);
  k_f32_to_bf16_T<<<(F_IN * FH + 255) / 256, 256, 0, stream>>>(res_w, RT, F_IN, FH);

  int gemm_blocks = (N_NODES / 16) * (FH / 16) / 8;  // 512 blocks x 8 waves
  k_wmma_gemm_bf16<<<gemm_blocks, 256, 0, stream>>>(Abf, WT, h, N_NODES, FH, F_IN);
  k_wmma_gemm_bf16<<<gemm_blocks, 256, 0, stream>>>(Abf, RT, resid, N_NODES, FH, F_IN);

  k_scores<<<(N_NODES * NHEAD + 255) / 256, 256, 0, stream>>>(h, a, ssrc, sdst);
  k_colsum<<<FH, 256, 0, stream>>>(h, cs);
  k_gat_attn<<<N_NODES, 256, 0, stream>>>(adj, h, ssrc, sdst, cs, resid, bias, res_b, out);
}
